// SinkhornSetNet_50010599195048
// MI455X (gfx1250) — matrix operand order repaired and checked by reading, not compiled, not run on hardware
//
#include <hip/hip_runtime.h>

#define REG_F 0.1f
#define INV_REG 10.0f
#define EPS_F 1e-10f
#define NITER 30

#define NB 32
#define NSEQ 1024
#define DIN 128
#define HDIM 512
#define EDIM 256
#define ROWS (NB * NSEQ)          // 32768 rows per input
#define ROWTILES (ROWS / 16)      // 2048 tiles per input

typedef __attribute__((ext_vector_type(16))) __bf16 v16bf;
typedef __attribute__((ext_vector_type(8)))  float  v8f;

// ---------- WMMA fragment loaders (per CDNA5 ISA 7.12.2 layouts) ----------

// A-matrix 16x32 bf16: lane -> row M = lane&15; K chunks at kb and 16+kb, kb = (lane>>4)*8
__device__ __forceinline__ v16bf load_afrag_b16(const __bf16* p, int lane) {
  int kb = (lane >> 4) << 3;
  v16bf f;
#pragma unroll
  for (int j = 0; j < 8; ++j) {
    f[j]     = p[kb + j];
    f[j + 8] = p[16 + kb + j];
  }
  return f;
}

__device__ __forceinline__ v16bf load_afrag_f32(const float* p, int lane) {
  int kb = (lane >> 4) << 3;
  v16bf f;
#pragma unroll
  for (int j = 0; j < 8; ++j) {
    f[j]     = (__bf16)p[kb + j];
    f[j + 8] = (__bf16)p[16 + kb + j];
  }
  return f;
}

// B-matrix 32x16 bf16: lane -> col N = lane&15; K chunks at kb and 16+kb, kb = (lane>>4)*8
__device__ __forceinline__ v16bf load_bfrag_b16(const __bf16* p, int lane) {
  int kb = (lane >> 4) << 3;
  v16bf f;
#pragma unroll
  for (int j = 0; j < 8; ++j) {
    f[j]     = p[kb + j];
    f[j + 8] = p[16 + kb + j];
  }
  return f;
}

#define WMMA_BF16(a, b, c) \
  __builtin_amdgcn_wmma_f32_16x16x32_bf16(false, (a), false, (b), (short)0, (c), false, false)

// ---------- prep: transpose weights to bf16 ----------
__global__ void wprep_kernel(const float* __restrict__ W1, const float* __restrict__ W2,
                             __bf16* __restrict__ W1T, __bf16* __restrict__ W2T) {
  int idx = blockIdx.x * 256 + threadIdx.x;
  if (idx < DIN * HDIM) {                 // W1: (128,512) -> W1T: (512,128)
    int d = idx >> 9, h = idx & 511;
    W1T[h * DIN + d] = (__bf16)W1[idx];
  }
  int idx2 = idx - DIN * HDIM;
  if (idx2 >= 0 && idx2 < HDIM * EDIM) {  // W2: (512,256) -> W2T: (256,512)
    int h = idx2 >> 8, e = idx2 & 255;
    W2T[e * HDIM + h] = (__bf16)W2[idx2];
  }
}

// ---------- prep: mask -> float, lengths, u/v targets ----------
__global__ void minit_kernel(const unsigned char* __restrict__ mask,
                             float* __restrict__ mf, float* __restrict__ ut,
                             float* __restrict__ u, float* __restrict__ v) {
  __shared__ float red[256];
  int b = blockIdx.x;
  float s = 0.f;
  for (int n = threadIdx.x; n < NSEQ; n += 256) {
    float m = mask[b * NSEQ + n] ? 1.f : 0.f;
    mf[b * NSEQ + n] = m;
    s += m;
  }
  red[threadIdx.x] = s;
  __syncthreads();
  for (int off = 128; off; off >>= 1) {
    if (threadIdx.x < off) red[threadIdx.x] += red[threadIdx.x + off];
    __syncthreads();
  }
  float len = fmaxf(red[0], 1.f);
  for (int n = threadIdx.x; n < NSEQ; n += 256) {
    float t = mf[b * NSEQ + n] / len;
    ut[b * NSEQ + n] = t;
    u[b * NSEQ + n]  = t;
    v[b * NSEQ + n]  = t;
  }
}

// ---------- encoder: x -> relu(xW1+b1) -> (.)W2+b2 -> l2norm -> bf16 emb ----------
// one wave per 16-row tile; LDS: per-wave 16x512 bf16 h-tile + 16x256 f32 e-tile
__global__ void encode_kernel(const float* __restrict__ Xa, const float* __restrict__ Xb,
                              const __bf16* __restrict__ W1T, const float* __restrict__ b1,
                              const __bf16* __restrict__ W2T, const float* __restrict__ b2,
                              __bf16* __restrict__ Aemb, __bf16* __restrict__ Bemb) {
  extern __shared__ char smem[];
  int wave = threadIdx.x >> 5;
  int lane = threadIdx.x & 31;
  __bf16* hbuf = (__bf16*)(smem) + (size_t)wave * 16 * HDIM;                    // 16 KB / wave
  float*  ebuf = (float*)(smem + 8 * 16 * HDIM * 2) + (size_t)wave * 16 * EDIM; // 16 KB / wave

  int tile  = blockIdx.x * 8 + wave;      // 4096 tiles: [0,2048)=A, [2048,4096)=B
  int which = tile >> 11;
  int rt    = tile & (ROWTILES - 1);
  const float* X = which ? Xb : Xa;
  __bf16* Emb    = which ? Bemb : Aemb;
  int row0 = rt * 16;
  int l16  = lane & 15;
  int half = lane >> 4;

  // A fragments for the 16x128 input row block (4 k-tiles of 32)
  v16bf xfrag[4];
#pragma unroll
  for (int kt = 0; kt < 4; ++kt)
    xfrag[kt] = load_afrag_f32(X + (size_t)(row0 + l16) * DIN + kt * 32, lane);

  // layer 1: 32 h-tiles of 16 cols each
  for (int ht = 0; ht < 32; ++ht) {
    float bias = b1[ht * 16 + l16];
    v8f acc;
#pragma unroll
    for (int r = 0; r < 8; ++r) acc[r] = bias;
#pragma unroll
    for (int kt = 0; kt < 4; ++kt) {
      v16bf bf = load_bfrag_b16(W1T + (size_t)(ht * 16 + l16) * DIN + kt * 32, lane);
      acc = WMMA_BF16(xfrag[kt], bf, acc);
    }
#pragma unroll
    for (int r = 0; r < 8; ++r) {
      int rr = r + half * 8;
      float hv = acc[r] > 0.f ? acc[r] : 0.f;  // ReLU
      hbuf[(size_t)rr * HDIM + ht * 16 + l16] = (__bf16)hv;
    }
  }
  __syncthreads();

  // layer 2: 16 e-tiles, K = 512 = 16 k-tiles
  for (int et = 0; et < 16; ++et) {
    float bias = b2[et * 16 + l16];
    v8f acc;
#pragma unroll
    for (int r = 0; r < 8; ++r) acc[r] = bias;
#pragma unroll
    for (int kt = 0; kt < 16; ++kt) {
      v16bf af = load_afrag_b16(hbuf + (size_t)l16 * HDIM + kt * 32, lane);
      v16bf bf = load_bfrag_b16(W2T + (size_t)(et * 16 + l16) * HDIM + kt * 32, lane);
      acc = WMMA_BF16(af, bf, acc);
    }
#pragma unroll
    for (int r = 0; r < 8; ++r) {
      int rr = r + half * 8;
      ebuf[(size_t)rr * EDIM + et * 16 + l16] = acc[r];
    }
  }
  __syncthreads();

  // l2 norm: row = lane&15, each half-wave covers 128 of 256 cols
  int row = l16;
  int c0  = half * 128;
  float s = 0.f;
  for (int c = 0; c < 128; ++c) {
    float e = ebuf[(size_t)row * EDIM + c0 + c];
    s += e * e;
  }
  s += __shfl_xor(s, 16, 32);                 // combine halves (same row)
  float scale = 1.f / fmaxf(sqrtf(s), 1e-12f);
  for (int c = 0; c < 128; ++c) {
    Emb[(size_t)(row0 + row) * EDIM + c0 + c] =
        (__bf16)(ebuf[(size_t)row * EDIM + c0 + c] * scale);
  }
}

// ---------- similarity + Gibbs kernel: K = exp((sim-1)/REG) * mask outer ----------
// one wave computes a 32x64 tile of K (8 accumulators); 64 WMMAs over E=256.
// A-frags reused across 4 column tiles, B-frags across 2 row tiles.
__global__ void simk_kernel(const __bf16* __restrict__ Aemb, const __bf16* __restrict__ Bemb,
                            const float* __restrict__ mf, float* __restrict__ K) {
  int wave = threadIdx.x >> 5;
  int lane = threadIdx.x & 31;
  int t    = blockIdx.x * 8 + wave;   // 32 batches * 32 row-tiles * 16 col-tiles
  int b    = t >> 9;
  int rem  = t & 511;
  int nt2  = rem >> 4;                // 32-row tile index
  int mt4  = rem & 15;                // 64-col tile index
  int row0 = nt2 * 32;
  int col0 = mt4 * 64;
  const __bf16* Ab = Aemb + (size_t)b * NSEQ * EDIM;
  const __bf16* Bb = Bemb + (size_t)b * NSEQ * EDIM;
  int l16  = lane & 15;
  int half = lane >> 4;

  v8f acc[8];
#pragma unroll
  for (int i = 0; i < 8; ++i)
#pragma unroll
    for (int r = 0; r < 8; ++r) acc[i][r] = 0.f;

#pragma unroll
  for (int kt = 0; kt < 8; ++kt) {
    v16bf a0 = load_afrag_b16(Ab + (size_t)(row0 + l16) * EDIM + kt * 32, lane);
    v16bf a1 = load_afrag_b16(Ab + (size_t)(row0 + 16 + l16) * EDIM + kt * 32, lane);
#pragma unroll
    for (int mi = 0; mi < 4; ++mi) {
      v16bf bf = load_bfrag_b16(Bb + (size_t)(col0 + mi * 16 + l16) * EDIM + kt * 32, lane);
      acc[mi]     = WMMA_BF16(a0, bf, acc[mi]);
      acc[4 + mi] = WMMA_BF16(a1, bf, acc[4 + mi]);
    }
  }

#pragma unroll
  for (int ni = 0; ni < 2; ++ni) {
#pragma unroll
    for (int mi = 0; mi < 4; ++mi) {
      float mcol = mf[b * NSEQ + col0 + mi * 16 + l16];
      float* Kp  = K + ((size_t)b * NSEQ + (size_t)(row0 + ni * 16)) * NSEQ + col0 + mi * 16;
#pragma unroll
      for (int r = 0; r < 8; ++r) {
        int rr = r + half * 8;
        float mrow = mf[b * NSEQ + row0 + ni * 16 + rr];
        float kv = expf((acc[ni * 4 + mi][r] - 1.f) * INV_REG) * mrow * mcol;
        Kp[(size_t)rr * NSEQ + l16] = kv;
      }
    }
  }
}

// ---------- Sinkhorn: u = ut / (K v + eps), one wave per row ----------
__global__ void sink_row_kernel(const float* __restrict__ K, const float* __restrict__ v,
                                const float* __restrict__ ut, float* __restrict__ u) {
  int wave = threadIdx.x >> 5;
  int lane = threadIdx.x & 31;
  int row  = blockIdx.x * 8 + wave;       // 0..32767
  int b    = row >> 10;
  const float* Kr = K + (size_t)row * NSEQ;
  const float* vb = v + (size_t)b * NSEQ;
  float s = 0.f;
#pragma unroll 4
  for (int i = lane; i < NSEQ; i += 32) {
    __builtin_prefetch(Kr + i + 256, 0, 1);   // global_prefetch_b8: stream K from L2
    s += Kr[i] * vb[i];
  }
#pragma unroll
  for (int off = 16; off; off >>= 1) s += __shfl_xor(s, off, 32);
  if (lane == 0) u[row] = ut[row] / (s + EPS_F);
}

// ---------- Sinkhorn: v = vt / (K^T u + eps), block covers 32 cols ----------
__global__ void sink_col_kernel(const float* __restrict__ K, const float* __restrict__ u,
                                const float* __restrict__ ut, float* __restrict__ v) {
  __shared__ float part[8][32];
  int b  = blockIdx.x >> 5;               // 32 col-tiles per batch
  int m0 = (blockIdx.x & 31) * 32;
  int tc = threadIdx.x & 31;
  int tr = threadIdx.x >> 5;
  const float* Kb = K + (size_t)b * NSEQ * NSEQ;
  const float* ub = u + (size_t)b * NSEQ;
  float s = 0.f;
  for (int n = tr; n < NSEQ; n += 8) {
    __builtin_prefetch(Kb + (size_t)(n + 8) * NSEQ + m0 + tc, 0, 1);
    s += Kb[(size_t)n * NSEQ + m0 + tc] * ub[n];
  }
  part[tr][tc] = s;
  __syncthreads();
  if (tr == 0) {
    float t = 0.f;
#pragma unroll
    for (int r = 0; r < 8; ++r) t += part[r][tc];
    v[(size_t)b * NSEQ + m0 + tc] = ut[(size_t)b * NSEQ + m0 + tc] / (t + EPS_F);
  }
}

// ---------- cost: sum u*K*v*C with C = -REG*log(K); per-block partials ----------
__global__ void cost_kernel(const float* __restrict__ K, const float* __restrict__ u,
                            const float* __restrict__ v, float* __restrict__ partials) {
  __shared__ float red[256];
  int b  = blockIdx.x >> 6;               // 64 blocks per batch, 16 rows each
  int rb = blockIdx.x & 63;
  const float* Kb = K + (size_t)b * NSEQ * NSEQ;
  const float* ub = u + (size_t)b * NSEQ;
  const float* vb = v + (size_t)b * NSEQ;
  int tr = threadIdx.x >> 4;              // row in tile (0..15)
  int tq = threadIdx.x & 15;              // col stride group
  int n  = rb * 16 + tr;
  float un = ub[n];
  float s = 0.f;
  for (int m = tq; m < NSEQ; m += 16) {
    float k = Kb[(size_t)n * NSEQ + m];
    if (k > 0.f) {
      float C = -REG_F * logf(k);
      s += un * k * vb[m] * C;
    }
  }
  red[threadIdx.x] = s;
  __syncthreads();
  for (int off = 128; off; off >>= 1) {
    if (threadIdx.x < off) red[threadIdx.x] += red[threadIdx.x + off];
    __syncthreads();
  }
  if (threadIdx.x == 0) partials[blockIdx.x] = red[0];
}

__global__ void final_kernel(const float* __restrict__ partials, float* __restrict__ out) {
  int b = threadIdx.x;
  if (b < NB) {
    float s = 0.f;
    for (int i = 0; i < 64; ++i) s += partials[b * 64 + i];
    out[b] = s;
  }
}

// ---------- host-side launch ----------
extern "C" void kernel_launch(void* const* d_in, const int* in_sizes, int n_in,
                              void* d_out, int out_size, void* d_ws, size_t ws_size,
                              hipStream_t stream) {
  (void)in_sizes; (void)n_in; (void)out_size; (void)ws_size;
  const float*         Xa    = (const float*)d_in[0];
  const float*         Xb    = (const float*)d_in[1];
  const unsigned char* maskp = (const unsigned char*)d_in[2];
  const float*         W1    = (const float*)d_in[3];
  const float*         b1    = (const float*)d_in[4];
  const float*         W2    = (const float*)d_in[5];
  const float*         b2    = (const float*)d_in[6];
  float* out = (float*)d_out;

  size_t off = 0;
  auto wsAlloc = [&](size_t bytes) -> void* {
    void* p = (char*)d_ws + off;
    off += (bytes + 255) & ~(size_t)255;
    return p;
  };
  __bf16* W1T   = (__bf16*)wsAlloc((size_t)HDIM * DIN * 2);
  __bf16* W2T   = (__bf16*)wsAlloc((size_t)EDIM * HDIM * 2);
  __bf16* Aemb  = (__bf16*)wsAlloc((size_t)ROWS * EDIM * 2);
  __bf16* Bemb  = (__bf16*)wsAlloc((size_t)ROWS * EDIM * 2);
  float*  mf    = (float*)wsAlloc((size_t)ROWS * 4);
  float*  ut    = (float*)wsAlloc((size_t)ROWS * 4);
  float*  uvec  = (float*)wsAlloc((size_t)ROWS * 4);
  float*  vvec  = (float*)wsAlloc((size_t)ROWS * 4);
  float*  Kbuf  = (float*)wsAlloc((size_t)NB * NSEQ * NSEQ * 4);   // 134 MB, L2-resident
  float*  parts = (float*)wsAlloc((size_t)NB * 64 * 4);

  wprep_kernel<<<768, 256, 0, stream>>>(W1, W2, W1T, W2T);
  minit_kernel<<<NB, 256, 0, stream>>>(maskp, mf, ut, uvec, vvec);

  // 4096 row tiles / 8 waves per block; 256 KB dynamic LDS per block
  encode_kernel<<<512, 256, 8 * 16 * HDIM * 2 + 8 * 16 * EDIM * 4, stream>>>(
      Xa, Xb, W1T, b1, W2T, b2, Aemb, Bemb);

  // 32 batches * 512 (32x64) tiles / 8 waves per block
  simk_kernel<<<2048, 256, 0, stream>>>(Aemb, Bemb, mf, Kbuf);

  for (int it = 0; it < NITER; ++it) {
    sink_row_kernel<<<ROWS / 8, 256, 0, stream>>>(Kbuf, vvec, ut, uvec);
    sink_col_kernel<<<NB * 32, 256, 0, stream>>>(Kbuf, uvec, ut, vvec);
  }

  cost_kernel<<<NB * 64, 256, 0, stream>>>(Kbuf, uvec, vvec, parts);
  final_kernel<<<1, 256, 0, stream>>>(parts, out);
}
